// KeypointEmbeddingAttention_30915174597234
// MI455X (gfx1250) — compile-verified, matrix operand order checked by reading, not run on hardware
//
#include <hip/hip_runtime.h>

// ---------------------------------------------------------------------------
// Fused keypoint-transformer for MI455X (gfx1250, wave32, WMMA).
//  grid = 2048 blocks x 256 threads; block owns BM=16 batch tokens, S=67 kps.
//  All matmuls -> v_wmma_f32_16x16x32_f16. Entire network in LDS: only the
//  26MB input and 17MB output touch HBM (~1.9us floor @ 23.3 TB/s); the
//  workload is compute-bound (~44 G MACs), so attention fragments are built
//  with vectorized LDS loads and WMMAs are issued back-to-back.
// ---------------------------------------------------------------------------

#define TB_TOTAL 32768
#define C_IN     201
#define NK       67
#define E_       8
#define L_       12
#define FF_      16
#define OUT_     128
#define KOUT     536     // NK*E
#define BM       16
#define EPS_     1e-5f

typedef _Float16 h16;
typedef __attribute__((ext_vector_type(16))) _Float16 v16h;
typedef __attribute__((ext_vector_type(8)))  _Float16 v8h;
typedef __attribute__((ext_vector_type(4)))  _Float16 v4h;
typedef __attribute__((ext_vector_type(8)))  float    v8f;
typedef __attribute__((ext_vector_type(4)))  float    f32x4;

__device__ __forceinline__ v8f wmma_f16(v16h a, v16h b, v8f c) {
  // D = A(16x32 f16) * B(32x16 f16) + C(16x16 f32)
  return __builtin_amdgcn_wmma_f32_16x16x32_f16(
      /*neg_a=*/false, a, /*neg_b=*/false, b,
      /*c_mod=*/(short)0, c, /*reuse_a=*/false, /*reuse_b=*/false);
}

__global__ __launch_bounds__(256, 1)
void pose_xfmr_fused(const float* __restrict__ pose,     // [32768,201]
                     const float* __restrict__ embw,     // [8,3]
                     const float* __restrict__ embb,     // [8]
                     const float* __restrict__ ipw,      // [L,24,8]
                     const float* __restrict__ ipb,      // [L,24]
                     const float* __restrict__ aow,      // [L,8,8]
                     const float* __restrict__ aob,      // [L,8]
                     const float* __restrict__ f1w,      // [L,16,8]
                     const float* __restrict__ f1b,      // [L,16]
                     const float* __restrict__ f2w,      // [L,8,16]
                     const float* __restrict__ f2b,      // [L,8]
                     const float* __restrict__ g1, const float* __restrict__ b1,
                     const float* __restrict__ g2, const float* __restrict__ b2,
                     const float* __restrict__ outw,     // [128,536]
                     const float* __restrict__ outb,     // [128]
                     float* __restrict__ out)            // [32768,128]
{
  // LDS: 17152 + 68608 + 72896 + 2144 = 160800 B (<= 320KB/WG spec)
  __shared__ h16   Xs[NK][BM][E_];     // residual stream (f16), 16B rows
  __shared__ h16   QKV[NK][BM][32];    // 0..7 Q*0.5, 8..15 K, 16..31 VW(2 heads)
  __shared__ h16   SB[8][NK][68];      // per-wave score/prob buffer (8B rows)
  __shared__ float RS[8][NK];          // per-wave softmax 1/rowsum

  const int tid  = threadIdx.x;
  const int lane = tid & 31;
  const int wave = tid >> 5;
  const int nIdx = lane & 15;          // N-col / M-row index inside a tile
  const int hi   = lane >> 4;          // half-wave selector
  const int m0   = blockIdx.x * BM;

  // ------------------- Phase 0: keypoint embedding -------------------------
  {
    float ew[24];
#pragma unroll
    for (int i = 0; i < 24; ++i) ew[i] = embw[i];
    float eb[8];
#pragma unroll
    for (int e = 0; e < 8; ++e) eb[e] = embb[e];
    if (tid < BM) __builtin_prefetch(pose + (size_t)(m0 + tid) * C_IN, 0, 0);
    for (int idx = tid; idx < NK * BM * E_; idx += 256) {
      int s = idx >> 7, r = idx & 127, m = r >> 3, e = r & 7;
      const float* p = pose + (size_t)(m0 + m) * C_IN + s * 3;
      float v = eb[e] + ew[e*3+0]*p[0] + ew[e*3+1]*p[1] + ew[e*3+2]*p[2];
      Xs[s][m][e] = (h16)v;
    }
  }
  __syncthreads();

  // ------------------------- encoder layers --------------------------------
#pragma unroll 1
  for (int l = 0; l < L_; ++l) {
    const float* ipw_l = ipw + l*24*8;
    const float* ipb_l = ipb + l*24;
    const float* aow_l = aow + l*64;
    const float* aob_l = aob + l*8;
    const float* f1w_l = f1w + l*16*8;
    const float* f1b_l = f1b + l*16;
    const float* f2w_l = f2w + l*8*16;
    const float* f2b_l = f2b + l*8;

    // ---- per-lane weight fragments (B matrices, K=8 padded to 32) ----
    v16h Bqk = {};                       // cols 0..7: 0.5*Wq row n ; 8..15: Wk
    float sc = (nIdx < 8) ? 0.5f : 1.0f; // fold 1/sqrt(DH)=0.5 into Q
    if (lane < 16) {
#pragma unroll
      for (int j = 0; j < 8; ++j) Bqk[j] = (h16)(ipw_l[nIdx*8 + j] * sc);
    }
    float biasQK = ipb_l[nIdx] * sc;

    v16h Bvw = {};                       // fused V*Wo projection, col n = h*8+eo
    float biasVW = 0.f;
    {
      int hh = nIdx >> 3, eo = nIdx & 7;
      float wr[4];
#pragma unroll
      for (int d = 0; d < 4; ++d) wr[d] = aow_l[eo*8 + 4*hh + d];
      if (lane < 16) {
#pragma unroll
        for (int j = 0; j < 8; ++j) {
          float a = 0.f;
#pragma unroll
          for (int d = 0; d < 4; ++d) a += wr[d] * ipw_l[(16 + 4*hh + d)*8 + j];
          Bvw[j] = (h16)a;
        }
      }
#pragma unroll
      for (int d = 0; d < 4; ++d) biasVW += wr[d] * ipb_l[16 + 4*hh + d];
    }

    // ---- Phase A: fused QKV/VW projection, per keypoint s ----
#pragma unroll 1
    for (int s = wave; s < NK; s += 8) {
      v16h a = {};
      if (lane < 16) {
        v8h x = *(const v8h*)&Xs[s][nIdx][0];           // one ds_load_b128
#pragma unroll
        for (int j = 0; j < 8; ++j) a[j] = x[j];
      }
      v8f d0 = {}; d0 = wmma_f16(a, Bqk, d0);
      v8f d1 = {}; d1 = wmma_f16(a, Bvw, d1);
#pragma unroll
      for (int r = 0; r < 8; ++r) {
        int mr = r + 8*hi;
        QKV[s][mr][nIdx]      = (h16)(d0[r] + biasQK);
        QKV[s][mr][16 + nIdx] = (h16)(d1[r] + biasVW);
      }
    }
    __syncthreads();

    // ---- Phase B: attention. wave owns m = 2*wave, 2*wave+1 ----
    float aob_n = (nIdx < 8) ? aob_l[nIdx] : 0.f;
#pragma unroll 1
    for (int mi = 0; mi < 2; ++mi) {
      const int m = 2*wave + mi;
      v8f acc[5] = {};                       // O tiles, accumulated over heads
#pragma unroll 1
      for (int h = 0; h < 2; ++h) {
        // -- score B-fragments (K-matrix tiles), hoisted over si --
        v16h bfs[5];
#pragma unroll
        for (int sj = 0; sj < 5; ++sj) {
          v16h bf = {};
          int tcol = sj*16 + nIdx;
          if (lane < 16 && (sj < 4 || tcol < NK)) {
            v4h kk = *(const v4h*)&QKV[tcol][m][8 + 4*h]; // one ds_load_b64
#pragma unroll
            for (int j = 0; j < 4; ++j) bf[j] = kk[j];
          }
          bfs[sj] = bf;
        }
        // -- scores = (0.5*Q) K^T : per si, 5 back-to-back WMMAs --
#pragma unroll 1
        for (int si = 0; si < 5; ++si) {
          v16h af = {};
          int srow = si*16 + nIdx;
          if (lane < 16 && (si < 4 || srow < NK)) {
            v4h qq = *(const v4h*)&QKV[srow][m][4*h];
#pragma unroll
            for (int j = 0; j < 4; ++j) af[j] = qq[j];
          }
#pragma unroll
          for (int sj = 0; sj < 5; ++sj) {
            v8f d = {};
            d = wmma_f16(af, bfs[sj], d);
            int tcol = sj*16 + nIdx;
#pragma unroll
            for (int r = 0; r < 8; ++r) {
              int ss = si*16 + r + 8*hi;
              if ((si < 4 || ss < NK) && (sj < 4 || tcol < NK))
                SB[wave][ss][tcol] = (h16)d[r];
            }
          }
        }
        // -- softmax, lane-parallel over rows; 4-wide LDS accesses --
        for (int s = lane; s < NK; s += 32) {
          h16* row = &SB[wave][s][0];
          float mx = -1e30f;
#pragma unroll 1
          for (int t = 0; t < 64; t += 4) {
            v4h q = *(const v4h*)(row + t);
            mx = fmaxf(mx, fmaxf(fmaxf((float)q[0], (float)q[1]),
                                 fmaxf((float)q[2], (float)q[3])));
          }
          mx = fmaxf(mx, fmaxf((float)row[64],
                               fmaxf((float)row[65], (float)row[66])));
          float sum = 0.f;
#pragma unroll 1
          for (int t = 0; t < 64; t += 4) {
            v4h q = *(const v4h*)(row + t);
            float p0 = __expf((float)q[0] - mx), p1 = __expf((float)q[1] - mx);
            float p2 = __expf((float)q[2] - mx), p3 = __expf((float)q[3] - mx);
            sum += (p0 + p1) + (p2 + p3);
            v4h pq; pq[0]=(h16)p0; pq[1]=(h16)p1; pq[2]=(h16)p2; pq[3]=(h16)p3;
            *(v4h*)(row + t) = pq;
          }
#pragma unroll
          for (int t = 64; t < 67; ++t) {
            float p = __expf((float)row[t] - mx);
            sum += p;
            row[t] = (h16)p;
          }
          RS[wave][s] = 1.0f / sum;
        }
        // -- AV B-fragments (VW tiles), hoisted over si --
        v16h bfav[3];
#pragma unroll
        for (int c = 0; c < 3; ++c) {
          v16h bf = {};
          if (nIdx < 8) {
#pragma unroll
            for (int j = 0; j < 8; ++j) {
              int t = 32*c + 8*hi + j;
              if (c < 2 || t < NK) bf[j] = QKV[t][m][16 + 8*h + nIdx];
            }
#pragma unroll
            for (int j = 0; j < 8; ++j) {
              int t = 32*c + 16 + 8*hi + j;
              if (c < 2 || t < NK) bf[8+j] = QKV[t][m][16 + 8*h + nIdx];
            }
          }
          bfav[c] = bf;
        }
        // -- O += P @ VW_h : 3 chained WMMAs per si; 1/rowsum folded into A --
#pragma unroll 1
        for (int si = 0; si < 5; ++si) {
          int srow = si*16 + nIdx;
          float rcp = (si < 4 || srow < NK) ? RS[wave][srow] : 0.f;
#pragma unroll
          for (int c = 0; c < 3; ++c) {
            v16h af = {};
            if (si < 4 || srow < NK) {
              if (c < 2) {                       // interior: 4x ds_load_b64
                v4h p0 = *(const v4h*)&SB[wave][srow][32*c + 8*hi];
                v4h p1 = *(const v4h*)&SB[wave][srow][32*c + 8*hi + 4];
                v4h p2 = *(const v4h*)&SB[wave][srow][32*c + 16 + 8*hi];
                v4h p3 = *(const v4h*)&SB[wave][srow][32*c + 16 + 8*hi + 4];
#pragma unroll
                for (int j = 0; j < 4; ++j) {
                  af[j]    = (h16)((float)p0[j] * rcp);
                  af[4+j]  = (h16)((float)p1[j] * rcp);
                  af[8+j]  = (h16)((float)p2[j] * rcp);
                  af[12+j] = (h16)((float)p3[j] * rcp);
                }
              } else {                           // tail chunk t = 64..66
#pragma unroll
                for (int j = 0; j < 8; ++j) {
                  int t = 64 + 8*hi + j;
                  if (t < NK) af[j] = (h16)((float)SB[wave][srow][t] * rcp);
                }
              }
            }
            acc[si] = wmma_f16(af, bfav[c], acc[si]);
          }
        }
      } // heads
      // residual add: x += attn_out (both heads) + aob
#pragma unroll 1
      for (int si = 0; si < 5; ++si) {
#pragma unroll
        for (int r = 0; r < 8; ++r) {
          int ss = si*16 + r + 8*hi;
          if ((si < 4 || ss < NK) && nIdx < 8)
            Xs[ss][m][nIdx] = (h16)((float)Xs[ss][m][nIdx] + acc[si][r] + aob_n);
        }
      }
    } // mi
    __syncthreads();

    // ---- Phase C: LayerNorm 1 ----
    {
      float gg[8], bb[8];
#pragma unroll
      for (int e = 0; e < 8; ++e) { gg[e] = g1[l*8+e]; bb[e] = b1[l*8+e]; }
      for (int tk = tid; tk < NK*BM; tk += 256) {
        int s = tk >> 4, m = tk & 15;
        float v[8], mean = 0.f;
        v8h x = *(const v8h*)&Xs[s][m][0];
#pragma unroll
        for (int e = 0; e < 8; ++e) { v[e] = (float)x[e]; mean += v[e]; }
        mean *= 0.125f;
        float var = 0.f;
#pragma unroll
        for (int e = 0; e < 8; ++e) { float t = v[e]-mean; var += t*t; }
        float rstd = rsqrtf(var*0.125f + EPS_);
        v8h y;
#pragma unroll
        for (int e = 0; e < 8; ++e) y[e] = (h16)((v[e]-mean)*rstd*gg[e] + bb[e]);
        *(v8h*)&Xs[s][m][0] = y;
      }
    }
    __syncthreads();

    // ---- Phase D: FF1 + ReLU (hidden reuses QKV cols 0..15) ----
    {
      v16h Bf1 = {};
      if (lane < 16) {
#pragma unroll
        for (int j = 0; j < 8; ++j) Bf1[j] = (h16)f1w_l[nIdx*8 + j];
      }
      float biasF1 = f1b_l[nIdx];
#pragma unroll 1
      for (int s = wave; s < NK; s += 8) {
        v16h a = {};
        if (lane < 16) {
          v8h x = *(const v8h*)&Xs[s][nIdx][0];
#pragma unroll
          for (int j = 0; j < 8; ++j) a[j] = x[j];
        }
        v8f d = {}; d = wmma_f16(a, Bf1, d);
#pragma unroll
        for (int r = 0; r < 8; ++r)
          QKV[s][r + 8*hi][nIdx] = (h16)fmaxf(d[r] + biasF1, 0.f);
      }
    }
    __syncthreads();

    // ---- Phase E: FF2 (K=16) + residual ----
    {
      v16h Bf2 = {};
      if (nIdx < 8) {
#pragma unroll
        for (int j = 0; j < 8; ++j) Bf2[j] = (h16)f2w_l[nIdx*16 + 8*hi + j];
      }
      float biasF2 = (nIdx < 8) ? f2b_l[nIdx] : 0.f;
#pragma unroll 1
      for (int s = wave; s < NK; s += 8) {
        v16h a = {};
        {
          v8h x = *(const v8h*)&QKV[s][nIdx][8*hi];     // one ds_load_b128
#pragma unroll
          for (int j = 0; j < 8; ++j) a[j] = x[j];
        }
        v8f d = {}; d = wmma_f16(a, Bf2, d);
#pragma unroll
        for (int r = 0; r < 8; ++r) {
          int mr = r + 8*hi;
          if (nIdx < 8)
            Xs[s][mr][nIdx] = (h16)((float)Xs[s][mr][nIdx] + d[r] + biasF2);
        }
      }
    }
    __syncthreads();

    // ---- Phase F: LayerNorm 2 ----
    {
      float gg[8], bb[8];
#pragma unroll
      for (int e = 0; e < 8; ++e) { gg[e] = g2[l*8+e]; bb[e] = b2[l*8+e]; }
      for (int tk = tid; tk < NK*BM; tk += 256) {
        int s = tk >> 4, m = tk & 15;
        float v[8], mean = 0.f;
        v8h x = *(const v8h*)&Xs[s][m][0];
#pragma unroll
        for (int e = 0; e < 8; ++e) { v[e] = (float)x[e]; mean += v[e]; }
        mean *= 0.125f;
        float var = 0.f;
#pragma unroll
        for (int e = 0; e < 8; ++e) { float t = v[e]-mean; var += t*t; }
        float rstd = rsqrtf(var*0.125f + EPS_);
        v8h y;
#pragma unroll
        for (int e = 0; e < 8; ++e) y[e] = (h16)((v[e]-mean)*rstd*gg[e] + bb[e]);
        *(v8h*)&Xs[s][m][0] = y;
      }
    }
    __syncthreads();
  } // layers

  // ---- Phase G: output projection [16,536]@[536,128] + tanh ----
  // K-map makes each half-fragment a contiguous 8xf16 row of Xs -> b128 loads.
  {
    const int o = 16*wave + nIdx;           // wave owns one 16-col N-tile
    const float* wrow = outw + (size_t)o * KOUT;
    v8f acc = {};
#pragma unroll 1
    for (int c = 0; c < 17; ++c) {          // K = 536 -> 17 chunks of 32
      v16h a = {}, b = {};
      {
        v8h x0 = *(const v8h*)&Xs[4*c + hi][nIdx][0];
        f32x4 w0 = *(const f32x4*)(wrow + 32*c + 8*hi);
        f32x4 w1 = *(const f32x4*)(wrow + 32*c + 8*hi + 4);
#pragma unroll
        for (int j = 0; j < 4; ++j) {
          a[j]   = x0[j];      a[4+j] = x0[4+j];
          b[j]   = (h16)w0[j]; b[4+j] = (h16)w1[j];
        }
        if (c < 16 || hi == 0) {            // k >= 536 only for c==16,hi==1
          v8h x1 = *(const v8h*)&Xs[4*c + 2 + hi][nIdx][0];
          f32x4 w2 = *(const f32x4*)(wrow + 32*c + 16 + 8*hi);
          f32x4 w3 = *(const f32x4*)(wrow + 32*c + 16 + 8*hi + 4);
#pragma unroll
          for (int j = 0; j < 4; ++j) {
            a[8+j]  = x1[j];      a[12+j] = x1[4+j];
            b[8+j]  = (h16)w2[j]; b[12+j] = (h16)w3[j];
          }
        }
      }
      acc = wmma_f16(a, b, acc);
    }
    float ob = outb[o];
#pragma unroll
    for (int r = 0; r < 8; ++r) {
      int m = r + 8*hi;
      out[(size_t)(m0 + m) * OUT_ + o] = tanhf(acc[r] + ob);
    }
  }
}

extern "C" void kernel_launch(void* const* d_in, const int* in_sizes, int n_in,
                              void* d_out, int out_size, void* d_ws, size_t ws_size,
                              hipStream_t stream) {
  (void)in_sizes; (void)n_in; (void)out_size; (void)d_ws; (void)ws_size;
  const float* pose = (const float*)d_in[0];
  const float* embw = (const float*)d_in[1];
  const float* embb = (const float*)d_in[2];
  const float* ipw  = (const float*)d_in[3];
  const float* ipb  = (const float*)d_in[4];
  const float* aow  = (const float*)d_in[5];
  const float* aob  = (const float*)d_in[6];
  const float* f1w  = (const float*)d_in[7];
  const float* f1b  = (const float*)d_in[8];
  const float* f2w  = (const float*)d_in[9];
  const float* f2b  = (const float*)d_in[10];
  const float* g1   = (const float*)d_in[11];
  const float* b1   = (const float*)d_in[12];
  const float* g2   = (const float*)d_in[13];
  const float* b2   = (const float*)d_in[14];
  const float* outw = (const float*)d_in[15];
  const float* outb = (const float*)d_in[16];
  float* out = (float*)d_out;

  dim3 grid(TB_TOTAL / BM);   // 2048 workgroups
  dim3 block(256);            // 8 wave32
  pose_xfmr_fused<<<grid, block, 0, stream>>>(
      pose, embw, embb, ipw, ipb, aow, aob, f1w, f1b, f2w, f2b,
      g1, b1, g2, b2, outw, outb, out);
}
